// BiRNNLM_36575941492860
// MI455X (gfx1250) — compile-verified, matrix operand order checked by reading, not run on hardware
//
#include <hip/hip_runtime.h>
#include <hip/hip_bf16.h>
#include <math.h>

// ---- problem sizes (from reference) ----
#define VOCAB 50257
#define EMB   64
#define H     32
#define SEQ   256
#define BATCH 16
#define ROWS  (SEQ * BATCH)          // 4096 flattened (s,b) rows
#define H2    (2 * H)                // 64 = concat hidden
#define NT    ((VOCAB + 15) / 16)    // 3142 vocab column tiles of 16

typedef __attribute__((ext_vector_type(16))) _Float16 v16h;
typedef __attribute__((ext_vector_type(8)))  float    v8f;

// K-index striping inside a v16h operand (16-bit A 16x32 / B 32x16 layout):
// lane-half 0 (lanes 0-15):  half hi -> K = hi<8 ? hi   : hi+8   (K 0-7, 16-23)
// lane-half 1 (lanes 16-31): half hi -> K = +8 of the above       (K 8-15, 24-31)
__device__ __forceinline__ int kmap(int lh, int hi) {
    return ((hi < 8) ? hi : hi + 8) + lh * 8;
}

// ---------------------------------------------------------------------------
// Kernel 1: embedding gather + input-gate precompute  gx = x @ W_ih + b_ih
// grid = ROWS, block = 192 (threads 0..95 -> left GRU gates, 96..191 -> right)
// ---------------------------------------------------------------------------
__global__ void k_gx(const int* __restrict__ idx, const float* __restrict__ embed,
                     const float* __restrict__ Wl, const float* __restrict__ bl,
                     const float* __restrict__ Wr, const float* __restrict__ br,
                     float* __restrict__ gx /* [2][ROWS][96] */) {
    __shared__ float xs[EMB];
    int row = blockIdx.x;
    int t = threadIdx.x;
    if (t < EMB) {
        int tok = idx[row];
        xs[t] = embed[(size_t)tok * EMB + t];
    }
    __syncthreads();
    int dir = t / 96, g = t % 96;
    const float* W = dir ? Wr : Wl;
    const float* bb = dir ? br : bl;
    float acc = bb[g];
#pragma unroll 16
    for (int e = 0; e < EMB; ++e) acc += xs[e] * W[e * 96 + g];
    gx[((size_t)dir * ROWS + row) * 96 + g] = acc;
}

// ---------------------------------------------------------------------------
// Kernel 2: GRU recurrence (serial over SEQ), one block per direction.
// block = 512 threads: thread = (b<<5)|j  computes hidden unit j of batch b.
// ---------------------------------------------------------------------------
__global__ void k_gru(const float* __restrict__ gx,
                      const float* __restrict__ Wlhh, const float* __restrict__ blhh,
                      const float* __restrict__ Wrhh, const float* __restrict__ brhh,
                      float* __restrict__ hout /* [ROWS][H2] */) {
    __shared__ float Whh[H * 96];   // 12 KB
    __shared__ float bhh[96];
    __shared__ float hs[BATCH * H];
    int dir = blockIdx.x;
    int t = threadIdx.x;
    const float* Wg = dir ? Wrhh : Wlhh;
    const float* bg = dir ? brhh : blhh;
    for (int i = t; i < H * 96; i += 512) Whh[i] = Wg[i];
    if (t < 96) bhh[t] = bg[t];
    hs[t] = 0.0f;
    __syncthreads();

    int b = t >> 5, j = t & 31;
    for (int step = 0; step < SEQ; ++step) {
        int s = dir ? (SEQ - 1 - step) : step;
        const float* g = gx + ((size_t)dir * ROWS + s * BATCH + b) * 96;
        float hr = bhh[j], hz = bhh[32 + j], hn = bhh[64 + j];
        float hprev = hs[b * H + j];
#pragma unroll 8
        for (int k = 0; k < H; ++k) {
            float hk = hs[b * H + k];
            hr += hk * Whh[k * 96 + j];
            hz += hk * Whh[k * 96 + 32 + j];
            hn += hk * Whh[k * 96 + 64 + j];
        }
        float r  = 1.0f / (1.0f + __expf(-(g[j]      + hr)));
        float z  = 1.0f / (1.0f + __expf(-(g[32 + j] + hz)));
        float nn = tanhf(g[64 + j] + r * hn);
        float hnew = (1.0f - z) * nn + z * hprev;
        __syncthreads();
        hs[b * H + j] = hnew;
        __syncthreads();
        hout[((size_t)s * BATCH + b) * H2 + dir * H + j] = hnew;
    }
}

// ---------------------------------------------------------------------------
// Kernel 3: pack h rows into f16 WMMA A-operand layout.
// task id = ((rt*2 + khalf) * 32 + lane); tasks = 256*2*32 = 16384
// ---------------------------------------------------------------------------
__global__ void k_packA(const float* __restrict__ hout, _Float16* __restrict__ Apk) {
    int gtid = blockIdx.x * blockDim.x + threadIdx.x;
    if (gtid >= (ROWS / 16) * 2 * 32) return;
    int lane = gtid & 31;
    int unit = gtid >> 5;            // rt*2 + khalf
    int kh = unit & 1, rt = unit >> 1;
    int m = lane & 15, lh = lane >> 4;
    v16h v;
#pragma unroll
    for (int hi = 0; hi < 16; ++hi) {
        int k = kmap(lh, hi) + kh * 32;
        v[hi] = (_Float16)hout[((size_t)rt * 16 + m) * H2 + k];
    }
    *(v16h*)(Apk + (size_t)gtid * 16) = v;
}

// ---------------------------------------------------------------------------
// Kernel 4: pack rnn_out (H2 x VOCAB fp32, row-major) into f16 B-operand layout.
// task id = ((nt*2 + khalf) * 32 + lane); tasks = NT*2*32
// ---------------------------------------------------------------------------
__global__ void k_packB(const float* __restrict__ Wout, _Float16* __restrict__ Bpk) {
    int gtid = blockIdx.x * blockDim.x + threadIdx.x;
    if (gtid >= NT * 2 * 32) return;
    int lane = gtid & 31;
    int unit = gtid >> 5;            // nt*2 + khalf
    int kh = unit & 1, nt = unit >> 1;
    int n = nt * 16 + (lane & 15);
    int lh = lane >> 4;
    v16h v;
#pragma unroll
    for (int hi = 0; hi < 16; ++hi) {
        int k = kmap(lh, hi) + kh * 32;
        float w = (n < VOCAB) ? Wout[(size_t)k * VOCAB + n] : 0.0f;
        v[hi] = (_Float16)w;
    }
    *(v16h*)(Bpk + (size_t)gtid * 16) = v;
}

// ---------------------------------------------------------------------------
// Kernel 5: fused projection + log_softmax.
// grid = 256 row-tiles (16 rows each), block = 512 (16 waves).
// Pass 1: WMMA over all vocab tiles with flash-style online (max, sumexp).
// Pass 2: recompute WMMA, write  logit - (max + log(sumexp)).
// C layout: lane l, slot i -> row m = i + 8*(l>=16), col n = ntile*16 + (l&15).
// ---------------------------------------------------------------------------
__global__ __launch_bounds__(512)
void k_logits(const _Float16* __restrict__ Apk, const _Float16* __restrict__ Bpk,
              const float* __restrict__ bias, float* __restrict__ out) {
    __shared__ float s_max[16][16];
    __shared__ float s_sum[16][16];
    __shared__ float s_lse[16];

    int rt = blockIdx.x;
    int tid = threadIdx.x;
    int wave = tid >> 5, lane = tid & 31;
    int ncl = lane & 15;
    int mh = (lane >> 4) * 8;        // row base for this lane's slots

    v16h a0 = *(const v16h*)(Apk + (((size_t)rt * 2 + 0) * 32 + lane) * 16);
    v16h a1 = *(const v16h*)(Apk + (((size_t)rt * 2 + 1) * 32 + lane) * 16);

    float vmax[8], vsum[8];
#pragma unroll
    for (int i = 0; i < 8; ++i) { vmax[i] = -INFINITY; vsum[i] = 0.0f; }

    // ---- pass 1: online max / sum-exp over all vocab columns ----
    for (int nt = wave; nt < NT; nt += 16) {
        const _Float16* bp = Bpk + (((size_t)nt * 2) * 32 + lane) * 16;
        v16h b0 = *(const v16h*)(bp);
        v16h b1 = *(const v16h*)(bp + 32 * 16);
        if (nt + 16 < NT)
            __builtin_prefetch(Bpk + (((size_t)(nt + 16) * 2) * 32 + lane) * 16, 0, 3);
        v8f c = {};
        c = __builtin_amdgcn_wmma_f32_16x16x32_f16(false, a0, false, b0, (short)0, c, false, false);
        c = __builtin_amdgcn_wmma_f32_16x16x32_f16(false, a1, false, b1, (short)0, c, false, false);
        int n = nt * 16 + ncl;
        if (n < VOCAB) {
            float bv = bias[n];
#pragma unroll
            for (int i = 0; i < 8; ++i) {
                float cv = c[i] + bv;
                float mo = vmax[i];
                float mn = fmaxf(mo, cv);
                vsum[i] = vsum[i] * __expf(mo - mn) + __expf(cv - mn);
                vmax[i] = mn;
            }
        }
    }

    // ---- cross-lane reduce within each 16-lane half (cols of a row) ----
#pragma unroll
    for (int i = 0; i < 8; ++i) {
        float m = vmax[i], s = vsum[i];
#pragma unroll
        for (int d = 1; d < 16; d <<= 1) {
            float mo = __shfl_xor(m, d, 32);
            float so = __shfl_xor(s, d, 32);
            float mn = fmaxf(m, mo);
            s = s * __expf(m - mn) + so * __expf(mo - mn);
            m = mn;
        }
        vmax[i] = m; vsum[i] = s;
    }
    if (ncl == 0) {                  // lane 0 -> rows 0-7, lane 16 -> rows 8-15
#pragma unroll
        for (int i = 0; i < 8; ++i) { s_max[wave][mh + i] = vmax[i]; s_sum[wave][mh + i] = vsum[i]; }
    }
    __syncthreads();

    // ---- combine across the 16 waves ----
    if (tid < 16) {
        float m = s_max[0][tid], s = s_sum[0][tid];
        for (int w = 1; w < 16; ++w) {
            float mo = s_max[w][tid], so = s_sum[w][tid];
            float mn = fmaxf(m, mo);
            s = s * __expf(m - mn) + so * __expf(mo - mn);
            m = mn;
        }
        s_lse[tid] = m + __logf(s);
    }
    __syncthreads();

    float lse[8];
#pragma unroll
    for (int i = 0; i < 8; ++i) lse[i] = s_lse[mh + i];

    // ---- pass 2: recompute logits, write log-softmax ----
    for (int nt = wave; nt < NT; nt += 16) {
        const _Float16* bp = Bpk + (((size_t)nt * 2) * 32 + lane) * 16;
        v16h b0 = *(const v16h*)(bp);
        v16h b1 = *(const v16h*)(bp + 32 * 16);
        if (nt + 16 < NT)
            __builtin_prefetch(Bpk + (((size_t)(nt + 16) * 2) * 32 + lane) * 16, 0, 3);
        v8f c = {};
        c = __builtin_amdgcn_wmma_f32_16x16x32_f16(false, a0, false, b0, (short)0, c, false, false);
        c = __builtin_amdgcn_wmma_f32_16x16x32_f16(false, a1, false, b1, (short)0, c, false, false);
        int n = nt * 16 + ncl;
        if (n < VOCAB) {
            float bv = bias[n];
#pragma unroll
            for (int i = 0; i < 8; ++i) {
                size_t row = (size_t)rt * 16 + mh + i;
                out[row * VOCAB + n] = c[i] + bv - lse[i];
            }
        }
    }
}

// ---------------------------------------------------------------------------
// Launch. Workspace layout (bytes, all 32B-aligned from a 256B-aligned base):
//   [0, 3145728)          gx      : 2*4096*96 f32
//   [3145728, 4194304)    hbuf    : 4096*64 f32
//   [4194304, 4718592)    Apk     : 256*2*32*16 f16
//   [4718592, 11153408)   Bpk     : 3142*2*32*16 f16
// total ~10.6 MB.
// ---------------------------------------------------------------------------
extern "C" void kernel_launch(void* const* d_in, const int* in_sizes, int n_in,
                              void* d_out, int out_size, void* d_ws, size_t ws_size,
                              hipStream_t stream) {
    const int*   idx      = (const int*)  d_in[0];
    const float* embed    = (const float*)d_in[1];
    const float* rnn_out  = (const float*)d_in[2];
    const float* out_bias = (const float*)d_in[3];
    const float* Wl_ih    = (const float*)d_in[4];
    const float* Wl_hh    = (const float*)d_in[5];
    const float* bl_ih    = (const float*)d_in[6];
    const float* bl_hh    = (const float*)d_in[7];
    const float* Wr_ih    = (const float*)d_in[8];
    const float* Wr_hh    = (const float*)d_in[9];
    const float* br_ih    = (const float*)d_in[10];
    const float* br_hh    = (const float*)d_in[11];
    float* out = (float*)d_out;

    char* ws = (char*)d_ws;
    float*     gx   = (float*)(ws);
    float*     hbuf = (float*)(ws + 3145728);
    _Float16*  Apk  = (_Float16*)(ws + 4194304);
    _Float16*  Bpk  = (_Float16*)(ws + 4718592);

    // 1) embedding + input gates
    k_gx<<<ROWS, 192, 0, stream>>>(idx, embed, Wl_ih, bl_ih, Wr_ih, br_ih, gx);
    // 2) bidirectional GRU recurrence (one block per direction)
    k_gru<<<2, 512, 0, stream>>>(gx, Wl_hh, bl_hh, Wr_hh, br_hh, hbuf);
    // 3) pack A (h tiles) into WMMA operand layout
    {
        int tasks = (ROWS / 16) * 2 * 32;
        k_packA<<<(tasks + 255) / 256, 256, 0, stream>>>(hbuf, Apk);
    }
    // 4) pack B (vocab weights) into WMMA operand layout, fp32 -> f16
    {
        int tasks = NT * 2 * 32;
        k_packB<<<(tasks + 255) / 256, 256, 0, stream>>>(rnn_out, Bpk);
    }
    // 5) fused projection + log_softmax (WMMA, two passes, output bound)
    k_logits<<<ROWS / 16, 512, 0, stream>>>(Apk, Bpk, out_bias, out);
}
